// MultiScaleDeformableAttention_8924942041779
// MI455X (gfx1250) — matrix operand order repaired and checked by reading, not compiled, not run on hardware
//
#include <hip/hip_runtime.h>
#include <math.h>

// ---------------------------------------------------------------------------
// MI455X (gfx1250) bf16-WMMA flash-attention.
// V chunks staged global->LDS by the Tensor Data Mover (tensor_load_to_lds),
// double-buffered and overlapped with WMMA compute; TENSORcnt-synchronized.
// ---------------------------------------------------------------------------

typedef __attribute__((ext_vector_type(16))) __bf16 v16bf;
typedef __attribute__((ext_vector_type(8)))  __bf16 v8bf;
typedef __attribute__((ext_vector_type(8)))  float  v8f;
typedef __attribute__((ext_vector_type(4)))  float  v4f;
typedef __attribute__((ext_vector_type(4)))  unsigned v4u;
typedef __attribute__((ext_vector_type(8)))  int      v8i;
typedef __attribute__((ext_vector_type(4)))  int      v4i;

#define DEV static __device__ __forceinline__

static constexpr int BATCH = 2;
static constexpr int LQ    = 4096;
static constexpr int LV    = 8192;
static constexpr int DM    = 256;

static constexpr int WAVES     = 8;    // waves per attention block
static constexpr int KC        = 64;   // keys per flash chunk
static constexpr int VRM_PITCH = 264;  // 256 + 8 pad (TDM pad: 4 DW / 512 B)
static constexpr int VT_PITCH  = 72;   // 64 + 8 pad  (TDM pad: 4 DW / 128 B)
static constexpr int P_PITCH   = 72;
static constexpr int O_PITCH   = 264;
// Per V-buffer (bf16 elems): vrm tile [64 keys][264] then vt tile [256 e][72]
static constexpr int V_REGION  = KC * VRM_PITCH + DM * VT_PITCH;   // 35328
// Two V buffers (double buffer) + per-wave P tiles.
static constexpr int LDS_ELEMS = 2 * V_REGION + WAVES * 16 * P_PITCH; // 79872

DEV int lane_id() { return (int)(threadIdx.x & 31); }

DEV v16bf concat16(v8bf lo, v8bf hi) {
  v16bf r;
#pragma unroll
  for (int i = 0; i < 8; ++i) { r[i] = lo[i]; r[i + 8] = hi[i]; }
  return r;
}

// A fragment (16x32, 16-bit): lane holds row m = lane%16,
// K = {kb..kb+7, kb+16..kb+23}, kb = (lane/16)*8.  p at tile origin.
DEV v16bf load_a(const __bf16* p, int ld) {
  const int l = lane_id();
  const __bf16* r = p + (size_t)(l & 15) * ld + ((l >> 4) << 3);
  return concat16(*(const v8bf*)r, *(const v8bf*)(r + 16));
}

// B fragment (32x16, 16-bit): lane holds column n = lane%16,
// K = {kb..kb+7, kb+16..kb+23}, kb = (lane/16)*8.  Column n of B sourced from ROW n of p.
DEV v16bf load_b(const __bf16* p, int ld) {
  const int l = lane_id();
  const __bf16* r = p + (size_t)(l & 15) * ld + ((l >> 4) << 3);
  return concat16(*(const v8bf*)r, *(const v8bf*)(r + 16));
}

DEV v8f wmma_bf16(v16bf a, v16bf b, v8f c) {
  return __builtin_amdgcn_wmma_f32_16x16x32_bf16(
      false, a, false, b, (short)0, c, false, false);
}

DEV void store_bf16x8(__bf16* p, v8f v) {
  v8bf t;
#pragma unroll
  for (int i = 0; i < 8; ++i) t[i] = (__bf16)v[i];
  *(v8bf*)p = t;
}

DEV void wait_lds() { asm volatile("s_wait_dscnt 0" ::: "memory"); }

// Flat LDS pointer -> LDS byte offset (LDS aperture keeps the offset in the
// low 32 bits of the flat address; high dword is the aperture base).
DEV unsigned lds_byte_off(const void* p) { return (unsigned)(size_t)p; }

// TDM: issue a 2D tensor_load_to_lds.  tile_w elems per row, tile_h rows,
// row stride `stride` elems, 2-byte elements, LDS padding pad_amount+1 DWORDs
// every (1<<pad_interval)*8 bytes.  Issued by whichever wave executes it.
DEV void tdm_load_2d(unsigned lds_off, const void* gsrc,
                     unsigned tile_w, unsigned tile_h, unsigned stride,
                     unsigned pad_interval, unsigned pad_amount) {
  const unsigned long long ga = (unsigned long long)(size_t)gsrc;
  v4u g0;
  g0[0] = 1u;                                   // count=1, user descriptor
  g0[1] = lds_off;                              // D#.lds_addr (bytes)
  g0[2] = (unsigned)(ga & 0xffffffffu);         // D#.global_addr[31:0]
  g0[3] = (unsigned)((ga >> 32) & 0x01ffffffu)  // D#.global_addr[56:32]
          | (2u << 30);                         // D#.type = 2 ("image")
  v8i g1;
  g1[0] = (int)((1u << 16)                      // data_size = 1 (2 bytes)
                | (1u << 20)                    // pad_enable
                | (pad_interval << 22) | (pad_amount << 25));
  g1[1] = (int)((tile_w & 0xffffu) << 16);      // tensor_dim0[15:0] @ bit48
  g1[2] = (int)((tile_w >> 16) | ((tile_h & 0xffffu) << 16)); // td0 hi, td1 lo
  g1[3] = (int)((tile_h >> 16) | (tile_w << 16));             // td1 hi, tile_dim0
  g1[4] = (int)(tile_h & 0xffffu);              // tile_dim1 (tile_dim2 = 0)
  g1[5] = (int)stride;                          // tensor_dim0_stride[31:0]
  g1[6] = 0;                                    // stride hi, dim1_stride lo
  g1[7] = 0;
  const v4i gz4 = {0, 0, 0, 0};                 // 2D: groups 2/3 unused
  const v8i gz8 = {0, 0, 0, 0, 0, 0, 0, 0};
  __builtin_amdgcn_tensor_load_to_lds(g0, g1, gz4, gz4, gz8, 0);
}

// ---------------------------------------------------------------------------
__global__ __launch_bounds__(256) void cvt_kernel(const float* __restrict__ src,
                                                  __bf16* __restrict__ dst,
                                                  int n, float scale) {
  int i = blockIdx.x * blockDim.x + threadIdx.x;
  int stride = gridDim.x * blockDim.x;
  for (; i < n; i += stride) dst[i] = (__bf16)(src[i] * scale);
}

// ---------------------------------------------------------------------------
// v = value . Wv^T + bv  -> vrm[b][k][e] and vt[b][e][k], bf16. Each wave
// computes its 16x16 tile in BOTH orientations so both stores are b128s.
// ---------------------------------------------------------------------------
__global__ __launch_bounds__(256) void vproj_kernel(
    const __bf16* __restrict__ valb, const __bf16* __restrict__ wvb,
    const float* __restrict__ bv,
    __bf16* __restrict__ vrm, __bf16* __restrict__ vt) {
  const int w  = blockIdx.x * (blockDim.x >> 5) + (threadIdx.x >> 5);
  const int et = w & 15;
  const int kt = w >> 4;
  const int l  = lane_id();
  const int row0 = kt * 16;
  const int e0   = et * 16;

  v8f c1 = {};  // (k x e)
  v8f c2 = {};  // (e x k)
#pragma unroll
  for (int ks = 0; ks < DM / 32; ++ks) {
    v16bf av = load_a(valb + (size_t)row0 * DM + ks * 32, DM);
    v16bf bw = load_b(wvb + (size_t)e0 * DM + ks * 32, DM);
    c1 = wmma_bf16(av, bw, c1);
    v16bf aw = load_a(wvb + (size_t)e0 * DM + ks * 32, DM);
    v16bf bvv = load_b(valb + (size_t)row0 * DM + ks * 32, DM);
    c2 = wmma_bf16(aw, bvv, c2);
  }
  {  // c1 elem r: (k = row0 + r + 8*(l>>4), e = e0 + (l&15)) -> vt[b][e][k]
    const float bias = bv[e0 + (l & 15)];
#pragma unroll
    for (int r = 0; r < 8; ++r) c1[r] += bias;
    const int b  = row0 / LV;
    const int kl = (row0 % LV) + ((l >> 4) << 3);
    store_bf16x8(vt + ((size_t)b * DM + (e0 + (l & 15))) * LV + kl, c1);
  }
  {  // c2 elem r: (e = e0 + r + 8*(l>>4), k = row0 + (l&15)) -> vrm[row][e]
    v8f bias = *(const v8f*)(bv + e0 + ((l >> 4) << 3));
    c2 += bias;
    const int krow = row0 + (l & 15);
    store_bf16x8(vrm + (size_t)krow * DM + e0 + ((l >> 4) << 3), c2);
  }
}

// ---------------------------------------------------------------------------
// Flash attention + fused output projection; 8 waves/block share each 64-key
// V chunk, DMA'd into double-buffered LDS by the TDM and overlapped with
// the WMMA pipeline.
// ---------------------------------------------------------------------------
__global__ __launch_bounds__(256) void attn_kernel(
    const __bf16* __restrict__ qb, const __bf16* __restrict__ vrm,
    const __bf16* __restrict__ vt, const __bf16* __restrict__ wob,
    const float* __restrict__ bo, float* __restrict__ out) {
  __shared__ __align__(16) __bf16 lds[LDS_ELEMS];
  const int tid  = (int)threadIdx.x;
  const int wave = tid >> 5;
  const int l    = lane_id();
  const int w    = blockIdx.x * WAVES + wave;
  const int b    = w >> 8;        // LQ/16 = 256 q-tiles per batch
  const int q0   = (w & 255) * 16;

  const __bf16* qbase = qb + ((size_t)b * LQ + q0) * DM;
  const __bf16* vrmb  = vrm + (size_t)b * LV * DM;
  const __bf16* vtb   = vt + (size_t)b * DM * LV;
  __bf16* p_lds = lds + 2 * V_REGION + wave * 16 * P_PITCH;

  // DMA one 64-key chunk (vrm tile + vt tile) into V buffer `buf`.
  auto issue_chunk = [&](int buf, int kc) {
    __bf16* base = lds + buf * V_REGION;
    tdm_load_2d(lds_byte_off(base), vrmb + (size_t)kc * DM,
                /*tile_w=*/DM, /*tile_h=*/KC, /*stride=*/DM,
                /*pad_interval=*/6, /*pad_amount=*/3);   // pitch 264
    tdm_load_2d(lds_byte_off(base + KC * VRM_PITCH), vtb + kc,
                /*tile_w=*/KC, /*tile_h=*/DM, /*stride=*/LV,
                /*pad_interval=*/4, /*pad_amount=*/3);   // pitch 72
  };

  v8f o[16];
#pragma unroll
  for (int i = 0; i < 16; ++i) o[i] = (v8f){};
  float mrow = -INFINITY, lrow = 0.f;

  // Prologue: DMA chunk 0, wait, publish to all waves.
  if (wave == 0) {
    issue_chunk(0, 0);
    __builtin_amdgcn_s_wait_tensorcnt(0);
  }
  __syncthreads();

#pragma unroll 1
  for (int ci = 0; ci < LV / KC; ++ci) {
    const int kc = ci * KC;
    const bool has_next = (ci + 1) < (LV / KC);
    __bf16* vrm_lds = lds + (ci & 1) * V_REGION;
    __bf16* vt_lds  = vrm_lds + KC * VRM_PITCH;

    // Kick off DMA of the NEXT chunk into the other buffer (overlapped).
    if (wave == 0 && has_next) issue_chunk((ci + 1) & 1, kc + KC);
    if (has_next && kc + 2 * KC < LV) {  // warm L2 two chunks ahead
      __builtin_prefetch(vrmb + (size_t)(kc + 2 * KC) * DM + (size_t)tid * 128, 0, 1);
      __builtin_prefetch(vtb + (size_t)tid * LV + kc + 2 * KC, 0, 1);
    }

    // ---- S^T = V . Q^T  (keys x q); Q pre-scaled by 1/sqrt(D) ----
    v8f st[KC / 16];
#pragma unroll
    for (int t = 0; t < KC / 16; ++t) st[t] = (v8f){};
#pragma unroll
    for (int ds = 0; ds < DM / 32; ++ds) {
      v16bf bq = load_b(qbase + ds * 32, DM);
#pragma unroll
      for (int t = 0; t < KC / 16; ++t) {
        v16bf av = load_a(vrm_lds + (t * 16) * VRM_PITCH + ds * 32, VRM_PITCH);
        st[t] = wmma_bf16(av, bq, st[t]);
      }
    }
    // ---- online softmax over keys (q column = lane&15, dup across halves) --
    float cmax = -INFINITY;
#pragma unroll
    for (int t = 0; t < KC / 16; ++t)
#pragma unroll
      for (int r = 0; r < 8; ++r) cmax = fmaxf(cmax, st[t][r]);
    cmax = fmaxf(cmax, __shfl_xor(cmax, 16, 32));
    const float mnew  = fmaxf(mrow, cmax);
    const float alpha = __expf(mrow - mnew);
    float psum = 0.f;
#pragma unroll
    for (int t = 0; t < KC / 16; ++t)
#pragma unroll
      for (int r = 0; r < 8; ++r) {
        float p = __expf(st[t][r] - mnew);
        st[t][r] = p;
        psum += p;
      }
    psum += __shfl_xor(psum, 16, 32);
    lrow = lrow * alpha + psum;
    mrow = mnew;
#pragma unroll
    for (int i = 0; i < 16; ++i)
#pragma unroll
      for (int r = 0; r < 8; ++r) o[i][r] *= alpha;
    // ---- P -> per-wave LDS tile, row-major P[q][key] ----
#pragma unroll
    for (int t = 0; t < KC / 16; ++t)
      store_bf16x8(p_lds + (size_t)(l & 15) * P_PITCH + t * 16 + ((l >> 4) << 3),
                   st[t]);
    wait_lds();
    // ---- O^T += V^T . P^T : A = vt_lds rows (e), B cols(q) from P rows ----
#pragma unroll
    for (int ks = 0; ks < KC / 32; ++ks) {
      v16bf bp = load_b(p_lds + ks * 32, P_PITCH);
#pragma unroll
      for (int e = 0; e < 16; ++e) {
        v16bf av = load_a(vt_lds + (e * 16) * VT_PITCH + ks * 32, VT_PITCH);
        o[e] = wmma_bf16(av, bp, o[e]);
      }
    }

    // Next chunk's DMA must be complete (and this chunk's readers done)
    // before the next iteration touches the other buffer.
    if (wave == 0 && has_next) __builtin_amdgcn_s_wait_tensorcnt(0);
    __syncthreads();
  }

  // ---- epilogue: normalize, stage O^T -> per-wave LDS (reuse V buffers) ----
  __bf16* o_slab = lds + wave * 16 * O_PITCH;
  const float inv = 1.f / lrow;
#pragma unroll
  for (int e = 0; e < 16; ++e) {
    v8f t = o[e];
#pragma unroll
    for (int r = 0; r < 8; ++r) t[r] *= inv;
    store_bf16x8(o_slab + (size_t)(l & 15) * O_PITCH + e * 16 + ((l >> 4) << 3),
                 t);
  }
  wait_lds();
  // ---- out^T(f,q) = Wo . O ; +bo ; contiguous f32 b128 stores ----
#pragma unroll 1
  for (int f = 0; f < 16; ++f) {
    v8f acc = {};
#pragma unroll
    for (int ks = 0; ks < DM / 32; ++ks) {
      v16bf aw = load_a(wob + (size_t)(f * 16) * DM + ks * 32, DM);
      v16bf bfrag = load_b(o_slab + ks * 32, O_PITCH);
      acc = wmma_bf16(aw, bfrag, acc);
    }
    v8f bias = *(const v8f*)(bo + f * 16 + ((l >> 4) << 3));
    acc += bias;
    float* p = out + ((size_t)b * LQ + q0 + (l & 15)) * DM + f * 16 +
               ((l >> 4) << 3);
    v4f lo4, hi4;
#pragma unroll
    for (int r = 0; r < 4; ++r) { lo4[r] = acc[r]; hi4[r] = acc[r + 4]; }
    *(v4f*)p = lo4;
    *(v4f*)(p + 4) = hi4;
  }
}

// ---------------------------------------------------------------------------
extern "C" void kernel_launch(void* const* d_in, const int* in_sizes, int n_in,
                              void* d_out, int out_size, void* d_ws,
                              size_t ws_size, hipStream_t stream) {
  const float* query = (const float*)d_in[0];
  const float* value = (const float*)d_in[1];
  // d_in[2]: value_spatial_shapes (unused by reference forward path)
  const float* Wv = (const float*)d_in[3];
  const float* bv = (const float*)d_in[4];
  const float* Wo = (const float*)d_in[5];
  const float* bo = (const float*)d_in[6];
  float* out = (float*)d_out;

  __bf16* qb   = (__bf16*)d_ws;                       // [B][LQ][DM]   4 MiB
  __bf16* valb = qb + (size_t)BATCH * LQ * DM;        // [B*LV][DM]    8 MiB
  __bf16* wvb  = valb + (size_t)BATCH * LV * DM;      // [DM][DM]    128 KiB
  __bf16* wob  = wvb + (size_t)DM * DM;               // [DM][DM]    128 KiB
  __bf16* vrm  = wob + (size_t)DM * DM;               // [B][LV][DM]   8 MiB
  __bf16* vt   = vrm + (size_t)BATCH * LV * DM;       // [B][DM][LV]   8 MiB

  const float qscale = 1.0f / 16.0f;  // 1/sqrt(DM)
  cvt_kernel<<<2048, 256, 0, stream>>>(query, qb, BATCH * LQ * DM, qscale);
  cvt_kernel<<<2048, 256, 0, stream>>>(value, valb, BATCH * LV * DM, 1.0f);
  cvt_kernel<<<64, 256, 0, stream>>>(Wv, wvb, DM * DM, 1.0f);
  cvt_kernel<<<64, 256, 0, stream>>>(Wo, wob, DM * DM, 1.0f);

  // 16384 wave-tiles, 8 waves/block
  vproj_kernel<<<2048, 256, 0, stream>>>(valb, wvb, bv, vrm, vt);

  // 512 q-tile waves, 8 waves/block; TDM-fed double-buffered V chunks
  attn_kernel<<<64, 256, 0, stream>>>(qb, vrm, vt, wob, bo, out);
}
